// BypassStreamUnit_87368224735582
// MI455X (gfx1250) — matrix operand order repaired and checked by reading, not compiled, not run on hardware
//
#include <hip/hip_runtime.h>
#include <hip/hip_bf16.h>

// ---------------------------------------------------------------------------
// CDNA5 (gfx1250) DCNv4 block:
//   BN1 -> pw0 GEMM -> (raw reshape) -> {value, offset/mask} GEMMs ->
//   DCNv4 bilinear gather -> w_op GEMM -> (raw reshape) -> BN2 -> pw1 GEMM
//
// GEMMs use v_wmma_f32_16x16x32_f16 (f16 in, f32 accumulate).
//   - B staged TRANSPOSED in LDS (n-major) so every lane's fragment is two
//     contiguous 8-half runs -> 2x ds_load_b128, no scalar u16 packing.
//   - 64x64 block tile, 8 waves, each wave = 16x32 (2 accumulators, shared
//     A fragment) -> 8 WMMAs per wave, 4 chained per accumulator (K=128).
//   - BN folded into per-K affine on the B-tile f32->f16 conversion.
//   - bias fused into the WMMA epilogue store.
// ---------------------------------------------------------------------------

typedef __attribute__((ext_vector_type(16))) _Float16 v16h;
typedef __attribute__((ext_vector_type(8)))  _Float16 v8h;
typedef __attribute__((ext_vector_type(8)))  float    v8f;

union F16Frag { v16h v; v8h h[2]; _Float16 e[16]; };
union CFrag   { v8f  v; float e[8]; };
union H4      { _Float16 e[4]; unsigned long long u; };

// ---------------------------------------------------------------------------
// Fold eval-mode BatchNorm (g, b, m, v) -> (scale, shift), 128 channels.
// ---------------------------------------------------------------------------
__global__ void bn_fold_kernel(const float* __restrict__ g,
                               const float* __restrict__ b,
                               const float* __restrict__ m,
                               const float* __restrict__ v,
                               float* __restrict__ scale,
                               float* __restrict__ shift) {
  int i = threadIdx.x;           // blockDim.x == 128
  float s = g[i] * rsqrtf(v[i] + 1e-5f);
  scale[i] = s;
  shift[i] = b[i] - m[i] * s;
}

// ---------------------------------------------------------------------------
// Unified WMMA GEMM:  C[M x N] = A[M x K=128] * B[K=128 x N] (+bias[n])
// Block tile 64(M) x 64(N); N is runtime (guards for the N=32 om GEMM).
// ---------------------------------------------------------------------------
__global__ __launch_bounds__(256)
void gemm_wmma_kernel(const float* __restrict__ A,
                      const float* __restrict__ B,
                      float* __restrict__ C,
                      const float* __restrict__ bias,    // per-N or null
                      const float* __restrict__ bscale,  // per-K affine on B or null
                      const float* __restrict__ bshift,
                      int N, int lda, int ldb, int ldc,
                      long long aBS, long long bBS, long long cBS) {
  const int tid  = threadIdx.x;
  const int wave = tid >> 5;
  const int lane = tid & 31;
  const int m0 = blockIdx.y * 64;
  const int n0 = blockIdx.x * 64;

  const float* Ab = A + (size_t)blockIdx.z * aBS;
  const float* Bb = B + (size_t)blockIdx.z * bBS;
  float*       Cb = C + (size_t)blockIdx.z * cBS;

  __shared__ _Float16 sA [64 * 128];   // 16 KB: A tile, row-major [m][k], f16
  __shared__ _Float16 sBt[64 * 128];   // 16 KB: B tile TRANSPOSED [n][k], f16

  // ---- stage A: 64 rows x 128 cols (f32 -> f16), b128 in / b64 out ----
#pragma unroll
  for (int i = 0; i < 8; ++i) {
    int vdx  = tid + 256 * i;
    int row  = vdx >> 5;          // 0..63
    int col4 = vdx & 31;          // 0..31 (x4 floats)
    const float4 t = *reinterpret_cast<const float4*>(
        Ab + (size_t)(m0 + row) * lda + (col4 << 2));
    H4 h;
    h.e[0] = (_Float16)t.x; h.e[1] = (_Float16)t.y;
    h.e[2] = (_Float16)t.z; h.e[3] = (_Float16)t.w;
    *reinterpret_cast<unsigned long long*>(&sA[row * 128 + (col4 << 2)]) = h.u;
  }

  // ---- stage B transposed: each thread converts two 4(K)x4(N) blocks ----
  // global: 4 row-wise float4 loads (+ BN affine); LDS: 4 K-contiguous b64.
#pragma unroll
  for (int i = 0; i < 2; ++i) {
    int bid = tid + 256 * i;      // 0..511
    int k   = (bid & 31) << 2;    // 0..124 (k block)
    int n   = (bid >> 5) << 2;    // 0..60  (n block within tile)
    const int gn = n0 + n;
    float4 r[4];
    if (gn < N) {
#pragma unroll
      for (int j = 0; j < 4; ++j) {
        r[j] = *reinterpret_cast<const float4*>(Bb + (size_t)(k + j) * ldb + gn);
        if (bscale) {
          const float s = bscale[k + j];
          const float h = bshift[k + j];
          r[j].x = r[j].x * s + h; r[j].y = r[j].y * s + h;
          r[j].z = r[j].z * s + h; r[j].w = r[j].w * s + h;
        }
      }
    } else {
#pragma unroll
      for (int j = 0; j < 4; ++j) r[j] = make_float4(0.f, 0.f, 0.f, 0.f);
    }
    // column c of the 4x4 block -> K-contiguous half4 at sBt[(n+c)][k]
    H4 c0, c1, c2, c3;
    c0.e[0] = (_Float16)r[0].x; c0.e[1] = (_Float16)r[1].x;
    c0.e[2] = (_Float16)r[2].x; c0.e[3] = (_Float16)r[3].x;
    c1.e[0] = (_Float16)r[0].y; c1.e[1] = (_Float16)r[1].y;
    c1.e[2] = (_Float16)r[2].y; c1.e[3] = (_Float16)r[3].y;
    c2.e[0] = (_Float16)r[0].z; c2.e[1] = (_Float16)r[1].z;
    c2.e[2] = (_Float16)r[2].z; c2.e[3] = (_Float16)r[3].z;
    c3.e[0] = (_Float16)r[0].w; c3.e[1] = (_Float16)r[1].w;
    c3.e[2] = (_Float16)r[2].w; c3.e[3] = (_Float16)r[3].w;
    *reinterpret_cast<unsigned long long*>(&sBt[(n + 0) * 128 + k]) = c0.u;
    *reinterpret_cast<unsigned long long*>(&sBt[(n + 1) * 128 + k]) = c1.u;
    *reinterpret_cast<unsigned long long*>(&sBt[(n + 2) * 128 + k]) = c2.u;
    *reinterpret_cast<unsigned long long*>(&sBt[(n + 3) * 128 + k]) = c3.u;
  }

  __syncthreads();

  // ---- per-wave 16(M) x 32(N): one shared A frag, two B frags per kt ----
  const int mi  = wave >> 1;      // 0..3 -> M sub-tile
  const int ni  = wave & 1;       // 0..1 -> N half (32 cols)
  const int tm  = mi * 16;
  const int tn0 = ni * 32;

  CFrag acc0, acc1;
#pragma unroll
  for (int r = 0; r < 8; ++r) { acc0.e[r] = 0.0f; acc1.e[r] = 0.0f; }

  const int mlane = lane & 15;
  const int khiA  = (lane >> 4) * 8;    // A: lanes 16-31 carry K+8 group
  const int khiB  = (lane >> 4) * 8;    // B: lanes 16-31 carry K+8 group

#pragma unroll
  for (int kt = 0; kt < 4; ++kt) {
    const int kbase = kt * 32;

    // A fragment (16x32, ISA 7.12.2): two contiguous 8-half runs.
    F16Frag af;
    const _Float16* ap = &sA[(tm + mlane) * 128 + kbase + khiA];
    af.h[0] = *reinterpret_cast<const v8h*>(ap);
    af.h[1] = *reinterpret_cast<const v8h*>(ap + 16);

    // B fragments (32x16): two contiguous 8-half runs in transposed LDS.
    F16Frag bf0, bf1;
    const _Float16* bp0 = &sBt[(tn0 + mlane) * 128 + kbase + khiB];
    const _Float16* bp1 = bp0 + 16 * 128;
    bf0.h[0] = *reinterpret_cast<const v8h*>(bp0);
    bf0.h[1] = *reinterpret_cast<const v8h*>(bp0 + 16);
    bf1.h[0] = *reinterpret_cast<const v8h*>(bp1);
    bf1.h[1] = *reinterpret_cast<const v8h*>(bp1 + 16);

    acc0.v = __builtin_amdgcn_wmma_f32_16x16x32_f16(
        false, af.v, false, bf0.v, (short)0, acc0.v, false, false);
    acc1.v = __builtin_amdgcn_wmma_f32_16x16x32_f16(
        false, af.v, false, bf1.v, (short)0, acc1.v, false, false);
  }

  // ---- epilogue: C/D layout (VGPR r -> M = r + 8*(lane>=16), N = lane&15) --
  const int mbase = m0 + tm + ((lane >> 4) << 3);
#pragma unroll
  for (int t = 0; t < 2; ++t) {
    const int ng = n0 + tn0 + t * 16 + mlane;
    if (ng < N) {
      const float bv = bias ? bias[ng] : 0.0f;
      const CFrag& a = t ? acc1 : acc0;
#pragma unroll
      for (int r = 0; r < 8; ++r) {
        Cb[(size_t)(mbase + r) * ldc + ng] = a.e[r] + bv;
      }
    }
  }
}

// ---------------------------------------------------------------------------
// DCNv4 core: one wave32 per output pixel, one float4 (4 channels) per lane.
//   value: [B*4096, 128], om: [B*4096, 32] ((dx,dy,m)*9), out: [B*4096, 128]
// Offsets/mask are wave-uniform -> validity branches are non-divergent.
// ---------------------------------------------------------------------------
__global__ __launch_bounds__(256)
void dcn_core_kernel(const float* __restrict__ value,
                     const float* __restrict__ om,
                     float* __restrict__ out) {
  const int wave = threadIdx.x >> 5;
  const int lane = threadIdx.x & 31;
  const int pix  = blockIdx.x * 8 + wave;      // 0..32767
  const int b    = pix >> 12;
  const int s    = pix & 4095;
  const int h    = s >> 6;
  const int w    = s & 63;

  const float* vb = value + ((size_t)b << 19);       // b * 4096 * 128
  const float* op = om + ((size_t)pix << 5);         // pix * 32

  float a0 = 0.f, a1 = 0.f, a2 = 0.f, a3 = 0.f;

#pragma unroll
  for (int k = 0; k < 9; ++k) {
    const float dx = op[k * 3 + 0];
    const float dy = op[k * 3 + 1];
    const float mk = op[k * 3 + 2];
    const float px = (float)(w + (k % 3) - 1) + dx;  // OFFSET_SCALE == 1
    const float py = (float)(h + (k / 3) - 1) + dy;
    const float fx0 = floorf(px);
    const float fy0 = floorf(py);
    const float fx = px - fx0;
    const float fy = py - fy0;
    const int ix = (int)fx0;
    const int iy = (int)fy0;
    const float wts[4] = { (1.f - fx) * (1.f - fy), fx * (1.f - fy),
                           (1.f - fx) * fy,         fx * fy };
#pragma unroll
    for (int c = 0; c < 4; ++c) {
      const int xi = ix + (c & 1);
      const int yi = iy + (c >> 1);
      const bool valid = (xi >= 0) & (xi < 64) & (yi >= 0) & (yi < 64);
      const float wc = wts[c] * mk;
      if (valid) {
        const float4 vv = *reinterpret_cast<const float4*>(
            vb + (((size_t)(yi * 64 + xi)) << 7) + (lane << 2));
        a0 += wc * vv.x; a1 += wc * vv.y;
        a2 += wc * vv.z; a3 += wc * vv.w;
      }
    }
  }

  float4 r; r.x = a0; r.y = a1; r.z = a2; r.w = a3;
  *reinterpret_cast<float4*>(out + (((size_t)pix) << 7) + (lane << 2)) = r;
}

// ---------------------------------------------------------------------------
// Launch: 6-kernel pipeline, same stream, workspace-reusing.
//   ws: [bn1 s/h | bn2 s/h | buf0 16MB | buf1 16MB | buf2 4MB]
//   buf0: Y1 (pw0 out)   -> later reused for DCN output
//   buf1: value          -> later reused for out2 (w_op result)
//   buf2: om
// ---------------------------------------------------------------------------
extern "C" void kernel_launch(void* const* d_in, const int* in_sizes, int n_in,
                              void* d_out, int out_size, void* d_ws, size_t ws_size,
                              hipStream_t stream) {
  const float* x      = (const float*)d_in[0];
  const float* bn1_g  = (const float*)d_in[1];
  const float* bn1_b  = (const float*)d_in[2];
  const float* bn1_m  = (const float*)d_in[3];
  const float* bn1_v  = (const float*)d_in[4];
  const float* w_pw0  = (const float*)d_in[5];
  const float* w_vp   = (const float*)d_in[6];
  const float* b_vp   = (const float*)d_in[7];
  const float* w_om   = (const float*)d_in[8];
  const float* b_om   = (const float*)d_in[9];
  const float* w_op   = (const float*)d_in[10];
  const float* bn2_g  = (const float*)d_in[11];
  const float* bn2_b  = (const float*)d_in[12];
  const float* bn2_m  = (const float*)d_in[13];
  const float* bn2_v  = (const float*)d_in[14];
  const float* w_pw1  = (const float*)d_in[15];
  float* out = (float*)d_out;

  float* ws   = (float*)d_ws;
  float* bn1s = ws + 0;
  float* bn1h = ws + 128;
  float* bn2s = ws + 256;
  float* bn2h = ws + 384;
  const size_t PB = (size_t)8 * 128 * 4096;   // 4,194,304 floats (16 MB)
  float* buf0 = ws + 512;                     // Y1, later DCN output
  float* buf1 = buf0 + PB;                    // value, later out2
  float* buf2 = buf1 + PB;                    // om [32768 x 32]

  const long long CHW = 128LL * 4096LL;       // 524288 per-batch stride

  // BN folds
  bn_fold_kernel<<<1, 128, 0, stream>>>(bn1_g, bn1_b, bn1_m, bn1_v, bn1s, bn1h);
  bn_fold_kernel<<<1, 128, 0, stream>>>(bn2_g, bn2_b, bn2_m, bn2_v, bn2s, bn2h);

  // K1: Y1[b][o][s] = sum_c w_pw0[o][c] * bn1(x[b][c][s])   (per batch)
  gemm_wmma_kernel<<<dim3(64, 2, 8), 256, 0, stream>>>(
      w_pw0, x, buf0, nullptr, bn1s, bn1h,
      4096, 128, 4096, 4096, 0LL, CHW, CHW);

  // Raw reshape: Y1 buffer is now X2 = [32768 x 128] row-major.
  // K2a: value = X2 @ w_vp + b_vp  -> buf1 [32768 x 128]
  gemm_wmma_kernel<<<dim3(2, 512, 1), 256, 0, stream>>>(
      buf0, w_vp, buf1, b_vp, nullptr, nullptr,
      128, 128, 128, 128, 0LL, 0LL, 0LL);

  // K2b: om = X2 @ w_om + b_om    -> buf2 [32768 x 32]
  gemm_wmma_kernel<<<dim3(1, 512, 1), 256, 0, stream>>>(
      buf0, w_om, buf2, b_om, nullptr, nullptr,
      32, 128, 32, 32, 0LL, 0LL, 0LL);

  // K3: DCNv4 bilinear gather -> buf0 (Y1 no longer needed) [32768 x 128]
  dcn_core_kernel<<<4096, 256, 0, stream>>>(buf1, buf2, buf0);

  // K4: out2 = DCN @ w_op -> buf1 (value no longer needed) [32768 x 128]
  gemm_wmma_kernel<<<dim3(2, 512, 1), 256, 0, stream>>>(
      buf0, w_op, buf1, nullptr, nullptr, nullptr,
      128, 128, 128, 128, 0LL, 0LL, 0LL);

  // Raw reshape back: buf1 is now (B, C, HW).
  // K5: out[b][o][s] = sum_c w_pw1[o][c] * bn2(out2[b][c][s])  (per batch)
  gemm_wmma_kernel<<<dim3(64, 2, 8), 256, 0, stream>>>(
      w_pw1, buf1, out, nullptr, bn2s, bn2h,
      4096, 128, 4096, 4096, 0LL, CHW, CHW);
}
